// MoELayer_69561290326687
// MI455X (gfx1250) — matrix operand order repaired and checked by reading, not compiled, not run on hardware
//
#include <hip/hip_runtime.h>
#include <hip/hip_bf16.h>
#include <math.h>

#define NUM_TOKENS 4096
#define HIDDEN     1024
#define INTER      2048
#define NUM_EXPERTS 16
#define ICHUNK     512      // INTER processed in 4 chunks of 512
#define NCHUNK     (INTER / ICHUNK)
#define LDSPAD     8        // bf16 elements of row padding (16B) -> conflict-free ds reads

typedef __attribute__((ext_vector_type(16))) __bf16 v16bf;
typedef __attribute__((ext_vector_type(8)))  float  v8f;

// ---------------------------------------------------------------------------
// Kernel 1: zero full_expert_outputs region + per-expert counters
// ---------------------------------------------------------------------------
__global__ void moe_zero_kernel(float* __restrict__ feo, int* __restrict__ counts) {
    const size_t n4 = (size_t)NUM_TOKENS * NUM_EXPERTS * HIDDEN / 4;
    float4* p = (float4*)feo;
    size_t i = (size_t)blockIdx.x * blockDim.x + threadIdx.x;
    size_t stride = (size_t)gridDim.x * blockDim.x;
    float4 z = make_float4(0.f, 0.f, 0.f, 0.f);
    for (; i < n4; i += stride) p[i] = z;
    if (blockIdx.x == 0 && threadIdx.x < NUM_EXPERTS) counts[threadIdx.x] = 0;
}

// ---------------------------------------------------------------------------
// Kernel 2: router — logits, top-2 + softmax, per-expert token lists
// One wave32 per token: lane l -> expert l&15, half (l>>4) of K.
// ---------------------------------------------------------------------------
__global__ void moe_router_kernel(const float* __restrict__ x,
                                  const float* __restrict__ gw,
                                  float* __restrict__ logits,
                                  int* __restrict__ sel, float* __restrict__ wt,
                                  int* __restrict__ counts, int* __restrict__ tlist) {
    int wave = threadIdx.x >> 5;
    int lane = threadIdx.x & 31;
    int token = blockIdx.x * (blockDim.x >> 5) + wave;
    if (token >= NUM_TOKENS) return;

    int e = lane & 15, half = lane >> 4;
    const float4* xr = (const float4*)(x + (size_t)token * HIDDEN + half * 512);
    const float4* gr = (const float4*)(gw + (size_t)e * HIDDEN + half * 512);
    float s = 0.f;
#pragma unroll 4
    for (int i = 0; i < 128; ++i) {
        float4 a = xr[i], b = gr[i];
        s += a.x * b.x + a.y * b.y + a.z * b.z + a.w * b.w;
    }
    s += __shfl_down(s, 16, 32);                 // combine K halves (wave32)
    if (lane < 16) logits[(size_t)token * NUM_EXPERTS + e] = s;

    float l[16];
#pragma unroll
    for (int i = 0; i < 16; ++i) l[i] = __shfl(s, i, 32);

    if (lane == 0) {
        int i1 = 0; float v1 = l[0];
        for (int i = 1; i < 16; ++i) if (l[i] > v1) { v1 = l[i]; i1 = i; }   // first max (ties)
        int i2 = -1; float v2 = -3.402823466e38f;
        for (int i = 0; i < 16; ++i) if (i != i1 && l[i] > v2) { v2 = l[i]; i2 = i; }
        float p1 = 1.f / (1.f + expf(v2 - v1));  // softmax over [v1, v2]
        float p2 = 1.f - p1;
        sel[2 * token] = i1; sel[2 * token + 1] = i2;
        wt[2 * token] = p1;  wt[2 * token + 1] = p2;
        int q1 = atomicAdd(&counts[i1], 1); tlist[i1 * NUM_TOKENS + q1] = token;
        int q2 = atomicAdd(&counts[i2], 1); tlist[i2 * NUM_TOKENS + q2] = token;
    }
}

// ---------------------------------------------------------------------------
// WMMA fragment helpers (16x16x32 bf16, wave32; contiguous k order chosen here, same for A and B)
// A (16x32, 16-bit): lane m=L%16; lanes 0-15 hold K 0..15, lanes 16-31
// hold K 16..31  -> one contiguous 16-element run (matches B).
// B (32x16, 16-bit): lane n=L%16; lanes 0-15 hold K 0..15, lanes 16-31 K 16..31
// -> one contiguous 16-element run (64B of the f32 weight row).
// ---------------------------------------------------------------------------
__device__ __forceinline__ v16bf frag_a_lds(const __bf16* __restrict__ row, int k0, int lane) {
    int b0 = (lane >= 16) ? 16 : 0;
    const __bf16* p = row + k0 + b0;
    v16bf f;
#pragma unroll
    for (int j = 0; j < 8; ++j) f[j] = p[j];
#pragma unroll
    for (int j = 0; j < 8; ++j) f[8 + j] = p[8 + j];
    return f;
}

__device__ __forceinline__ v16bf frag_b_global_f32(const float* __restrict__ row, int k0, int lane) {
    int b0 = (lane >= 16) ? 16 : 0;
    const float4* p = (const float4*)(row + k0 + b0);
    v16bf f;
#pragma unroll
    for (int q = 0; q < 4; ++q) {
        float4 v = p[q];
        f[4 * q + 0] = (__bf16)v.x; f[4 * q + 1] = (__bf16)v.y;
        f[4 * q + 2] = (__bf16)v.z; f[4 * q + 3] = (__bf16)v.w;
    }
    return f;
}

// Branch-free tanh-form GELU: one hardware exp (TRANS), no divergence.
// 0.5*(1+tanh(u)) == sigmoid(2u); overflow-safe for |v| large.
__device__ __forceinline__ float gelu_fast(float v) {
    float u = v + 0.044715f * v * v * v;
    return v / (1.f + __expf(-1.5957691216057308f * u));
}

// ---------------------------------------------------------------------------
// Kernel 3: per-expert fused FFN: Y = gelu(X @ W1^T) @ W2^T, gathered tokens.
// grid = (256 row-tiles, 16 experts); 256 threads = 8 waves per block.
// k-outer loops: one shared A-fragment feeds 4 (phase1) / 8 (phase2) WMMAs.
// ---------------------------------------------------------------------------
__global__ __launch_bounds__(256) void moe_ffn_kernel(
        const float* __restrict__ w1, const float* __restrict__ w2,
        const float* __restrict__ x,  const int* __restrict__ counts,
        const int* __restrict__ tlist, float* __restrict__ feo) {
    __shared__ __bf16 Xs[16][HIDDEN + LDSPAD];   // 33 KB
    __shared__ __bf16 Hs[16][ICHUNK + LDSPAD];   // 16.6 KB
    __shared__ int    toks[16];

    const int e    = blockIdx.y;
    const int tile = blockIdx.x;
    const int cnt  = counts[e];
    const int row0 = tile * 16;
    if (row0 >= cnt) return;                      // uniform early exit
    const int nrows = min(16, cnt - row0);

    const int tid  = threadIdx.x;
    const int wave = tid >> 5;
    const int lane = tid & 31;
    const int ln16 = lane & 15;

    if (tid < 16) toks[tid] = (tid < nrows) ? tlist[e * NUM_TOKENS + row0 + tid] : 0;
    __syncthreads();

    // Gather X tile (f32 -> bf16) into LDS, coalesced float4 units
    for (int u = tid; u < 16 * (HIDDEN / 4); u += 256) {
        int r  = u / (HIDDEN / 4);
        int c4 = u % (HIDDEN / 4);
        float4 v = make_float4(0.f, 0.f, 0.f, 0.f);
        if (r < nrows) v = ((const float4*)(x + (size_t)toks[r] * HIDDEN))[c4];
        __bf16* dst = &Xs[r][c4 * 4];
        dst[0] = (__bf16)v.x; dst[1] = (__bf16)v.y;
        dst[2] = (__bf16)v.z; dst[3] = (__bf16)v.w;
    }
    __syncthreads();

    // per-wave w2 row pointers (8 output n-tiles over HIDDEN), fixed all kernel
    const float* brow2[8];
#pragma unroll
    for (int j = 0; j < 8; ++j) {
        int wrow = ((wave + j * 8) * 16 + ln16);          // w2 row = B column
        brow2[j] = w2 + ((size_t)e * HIDDEN + wrow) * INTER;
        __builtin_prefetch(brow2[j], 0, 1);               // global_prefetch_b8
    }

    v8f yacc[8];
#pragma unroll
    for (int j = 0; j < 8; ++j) yacc[j] = (v8f){};

    for (int h = 0; h < NCHUNK; ++h) {
        // ---- phase 1: H = gelu(X @ W1_chunk^T); 4 n-tiles per wave, k-outer
        const float* brow1[4];
#pragma unroll
        for (int t = 0; t < 4; ++t) {
            int n1   = wave + t * 8;                      // 0..31 within chunk
            int wrow = h * ICHUNK + n1 * 16 + ln16;       // w1 row = B column
            brow1[t] = w1 + ((size_t)e * INTER + wrow) * HIDDEN;
            __builtin_prefetch(brow1[t], 0, 1);
        }
        v8f acc1[4];
#pragma unroll
        for (int t = 0; t < 4; ++t) acc1[t] = (v8f){};

        for (int k0 = 0; k0 < HIDDEN; k0 += 32) {
            v16bf a = frag_a_lds(&Xs[ln16][0], k0, lane); // shared across 4 WMMAs
#pragma unroll
            for (int t = 0; t < 4; ++t) {
                v16bf b = frag_b_global_f32(brow1[t], k0, lane);
                acc1[t] = __builtin_amdgcn_wmma_f32_16x16x32_bf16(
                              false, a, false, b, (short)0, acc1[t], false, false);
            }
        }
        {
            int m0 = (lane >= 16) ? 8 : 0;
#pragma unroll
            for (int t = 0; t < 4; ++t) {
                int nn = (wave + t * 8) * 16 + ln16;
#pragma unroll
                for (int r = 0; r < 8; ++r)
                    Hs[m0 + r][nn] = (__bf16)gelu_fast(acc1[t][r]);
            }
        }
        __syncthreads();

        // ---- phase 2: Y += H @ W2_chunk^T; 8 n-tiles per wave, k-outer
        for (int k0 = 0; k0 < ICHUNK; k0 += 32) {
            v16bf a = frag_a_lds(&Hs[ln16][0], k0, lane); // shared across 8 WMMAs
            int kk = h * ICHUNK + k0;
#pragma unroll
            for (int j = 0; j < 8; ++j) {
                v16bf b = frag_b_global_f32(brow2[j], kk, lane);
                yacc[j] = __builtin_amdgcn_wmma_f32_16x16x32_bf16(
                              false, a, false, b, (short)0, yacc[j], false, false);
            }
        }
        __syncthreads();                                  // before next chunk rewrites Hs
    }

    // ---- store: feo[token][e][col], C/D layout: (m = r + 8*(lane>=16), n = lane&15)
    int m0 = (lane >= 16) ? 8 : 0;
#pragma unroll
    for (int j = 0; j < 8; ++j) {
        int col = (wave + j * 8) * 16 + ln16;
#pragma unroll
        for (int r = 0; r < 8; ++r) {
            int m = m0 + r;
            if (row0 + m < cnt) {
                feo[((size_t)toks[m] * NUM_EXPERTS + e) * HIDDEN + col] = yacc[j][r];
            }
        }
    }
}

// ---------------------------------------------------------------------------
// Kernel 4: final gather  final[n] = sum_k wt[n,k] * feo[n, sel[n,k], :]
// ---------------------------------------------------------------------------
__global__ void moe_gather_kernel(const float* __restrict__ feo,
                                  const int* __restrict__ sel,
                                  const float* __restrict__ wt,
                                  float* __restrict__ out) {
    int t = blockIdx.x;
    int e0 = sel[2 * t], e1 = sel[2 * t + 1];
    float p0 = wt[2 * t], p1 = wt[2 * t + 1];
    const float* f0 = feo + ((size_t)t * NUM_EXPERTS + e0) * HIDDEN;
    const float* f1 = feo + ((size_t)t * NUM_EXPERTS + e1) * HIDDEN;
    float* o = out + (size_t)t * HIDDEN;
    for (int c = threadIdx.x; c < HIDDEN; c += blockDim.x)
        o[c] = p0 * f0[c] + p1 * f1[c];
}

// ---------------------------------------------------------------------------
extern "C" void kernel_launch(void* const* d_in, const int* in_sizes, int n_in,
                              void* d_out, int out_size, void* d_ws, size_t ws_size,
                              hipStream_t stream) {
    const float* x  = (const float*)d_in[0];   // [4096,1024]
    const float* gw = (const float*)d_in[1];   // [16,1024]
    const float* w1 = (const float*)d_in[2];   // [16,2048,1024]
    const float* w2 = (const float*)d_in[3];   // [16,1024,2048]

    float* out_final  = (float*)d_out;                                   // [4096,1024]
    float* out_feo    = out_final + (size_t)NUM_TOKENS * HIDDEN;         // [4096,16,1024]
    float* out_logits = out_feo + (size_t)NUM_TOKENS * NUM_EXPERTS * HIDDEN; // [4096,16]

    // workspace layout
    int*   counts = (int*)d_ws;                         // 16 (padded to 64)
    int*   sel    = counts + 64;                        // 4096*2 ints
    float* wt     = (float*)(sel + 2 * NUM_TOKENS);     // 4096*2 floats
    int*   tlist  = (int*)(wt + 2 * NUM_TOKENS);        // 16*4096 ints

    moe_zero_kernel<<<4096, 256, 0, stream>>>(out_feo, counts);
    moe_router_kernel<<<NUM_TOKENS / 8, 256, 0, stream>>>(x, gw, out_logits,
                                                          sel, wt, counts, tlist);
    dim3 ffn_grid(NUM_TOKENS / 16, NUM_EXPERTS);
    moe_ffn_kernel<<<ffn_grid, 256, 0, stream>>>(w1, w2, x, counts, tlist, out_feo);
    moe_gather_kernel<<<NUM_TOKENS, 256, 0, stream>>>(out_feo, sel, wt, out_final);
}
